// HierarchicalVoxelAttention_21715354649028
// MI455X (gfx1250) — matrix operand order repaired and checked by reading, not compiled, not run on hardware
//
#include <hip/hip_runtime.h>
#include <hip/hip_bf16.h>

// ---------------- problem constants ----------------
#define N_PART 500000
#define NHEAD  4

// hard voxel-count bounds: coords in [0,8)^3, grids {0.5,1.0,2.0}
#define MAXM0  4096
#define MAXM1  512
#define MAXM2  64
#define MAXM_TOT (MAXM0 + MAXM1 + MAXM2)   // 4672

// ---------------- workspace layout (float offsets) ----------------
#define OFF_MACRO  0
#define OFF_COUNTS (MAXM_TOT * 128)                    // 598016
#define OFF_X      (OFF_COUNTS + MAXM_TOT)             // 602688
#define OFF_QKV    (OFF_X + MAXM_TOT * 128)            // 1200704
#define OFF_ATTN   (OFF_QKV + MAXM0 * 384)             // 2773568
#define OFF_TMP    (OFF_ATTN + MAXM0 * 128)            // 3297856
#define OFF_FF     (OFF_TMP + MAXM0 * 128)             // 3822144
#define OFF_WFB    (OFF_FF + MAXM0 * 256)              // 4870720 (bf16 Wf: 128*512*2B)
#define WS_FLOATS  (OFF_WFB + 32768)

// LDS pitch for staged Wf: 512 + 8 bf16 -> row stride 1040B = 260 dwords,
// 260 % 64 = 4 banks/col step -> lanes 0..15 (b128 each) cover all 64 banks.
#define WF_PITCH 520
#define WF_LDS_BYTES (128 * WF_PITCH * 2)   // 133120

typedef __attribute__((ext_vector_type(2)))  float   v2f;
typedef __attribute__((ext_vector_type(8)))  float   v8f;
typedef __attribute__((ext_vector_type(8)))  __bf16  v8bf;
typedef __attribute__((ext_vector_type(16))) __bf16  v16bf;

// async-LDS builtin wants int-vector(4) pointers (per hipcc diagnostic)
typedef int v4i_async __attribute__((vector_size(16)));
typedef __attribute__((address_space(1))) v4i_async* gptr_v4i;
typedef __attribute__((address_space(3))) v4i_async* lptr_v4i;

#if defined(__has_builtin)
#if __has_builtin(__builtin_amdgcn_global_load_async_to_lds_b128) && \
    __has_builtin(__builtin_amdgcn_s_wait_asynccnt)
#define USE_ASYNC_LDS 1
#endif
#endif

// ---------------- wave32 reductions ----------------
__device__ __forceinline__ float wred_sum(float v) {
    v += __shfl_xor(v, 16, 32); v += __shfl_xor(v, 8, 32); v += __shfl_xor(v, 4, 32);
    v += __shfl_xor(v, 2, 32);  v += __shfl_xor(v, 1, 32);
    return v;
}
// half-wave (16-lane) reductions: xor masks < 16 never cross the lane-16 boundary
__device__ __forceinline__ float hred_sum(float v) {
    v += __shfl_xor(v, 8, 32); v += __shfl_xor(v, 4, 32);
    v += __shfl_xor(v, 2, 32); v += __shfl_xor(v, 1, 32);
    return v;
}
__device__ __forceinline__ float hred_max(float v) {
    v = fmaxf(v, __shfl_xor(v, 8, 32)); v = fmaxf(v, __shfl_xor(v, 4, 32));
    v = fmaxf(v, __shfl_xor(v, 2, 32)); v = fmaxf(v, __shfl_xor(v, 1, 32));
    return v;
}

// ---------------- utility kernels ----------------
__global__ void zero_kernel(float* __restrict__ p, int n) {
    int i = blockIdx.x * blockDim.x + threadIdx.x;
    if (i < n) p[i] = 0.0f;
}

__global__ void cvt_wf_kernel(const float* __restrict__ Wf, __bf16* __restrict__ o) {
    int i = blockIdx.x * blockDim.x + threadIdx.x;
    if (i < 128 * 512) o[i] = (__bf16)Wf[i];
}

// scatter-mean pooling: one wave per particle (lane = 4 channels), all 3 scales in one h pass
__global__ __launch_bounds__(256)
void scatter_kernel(const float* __restrict__ h, const int* __restrict__ i0,
                    const int* __restrict__ i1, const int* __restrict__ i2,
                    float* __restrict__ macro, float* __restrict__ counts) {
    int lane = threadIdx.x & 31;
    int gw = blockIdx.x * (blockDim.x >> 5) + (threadIdx.x >> 5);
    int nw = gridDim.x * (blockDim.x >> 5);
    for (int n = gw; n < N_PART; n += nw) {
        float4 v = ((const float4*)(h + (size_t)n * 128))[lane];
        int a = i0[n], b = i1[n], c = i2[n];
        float* p0 = macro + (size_t)a * 128 + lane * 4;
        float* p1 = macro + (size_t)(MAXM0 + b) * 128 + lane * 4;
        float* p2 = macro + (size_t)(MAXM0 + MAXM1 + c) * 128 + lane * 4;
        atomicAdd(p0 + 0, v.x); atomicAdd(p0 + 1, v.y); atomicAdd(p0 + 2, v.z); atomicAdd(p0 + 3, v.w);
        atomicAdd(p1 + 0, v.x); atomicAdd(p1 + 1, v.y); atomicAdd(p1 + 2, v.z); atomicAdd(p1 + 3, v.w);
        atomicAdd(p2 + 0, v.x); atomicAdd(p2 + 1, v.y); atomicAdd(p2 + 2, v.z); atomicAdd(p2 + 3, v.w);
        if (lane == 0) {
            atomicAdd(counts + a, 1.0f);
            atomicAdd(counts + MAXM0 + b, 1.0f);
            atomicAdd(counts + MAXM0 + MAXM1 + c, 1.0f);
        }
    }
}

__global__ void finalize_kernel(float* __restrict__ macro, const float* __restrict__ counts) {
    int i = blockIdx.x * blockDim.x + threadIdx.x;
    if (i < MAXM_TOT * 128) macro[i] /= fmaxf(counts[i >> 7], 1.0f);
}

// ---------------- fp32 WMMA GEMM: C[M,Nout] = A[M,K] @ W[Nout,K]^T + bias ----------------
__global__ __launch_bounds__(32)
void gemm_wmma_f32(const float* __restrict__ A, const float* __restrict__ W,
                   const float* __restrict__ bias, float* __restrict__ C,
                   const int* __restrict__ Mptr, int K, int Nout, int relu) {
    int M = *Mptr;
    int tm = blockIdx.x, tn = blockIdx.y;
    if (tm * 16 >= M) return;                    // uniform: EXEC stays all-ones
    int lane = threadIdx.x & 31, ln = lane & 15, hi = lane >> 4;
    int row = tm * 16 + ln; if (row >= M) row = M - 1;   // clamp, masked at store
    int col = tn * 16 + ln;
    const float* a_ = A + (size_t)row * K;
    const float* w_ = W + (size_t)col * K;
    v8f acc = {};
    for (int k = 0; k < K; k += 4) {
        v2f av, bv;
        av.x = a_[k + hi * 2]; av.y = a_[k + hi * 2 + 1];
        bv.x = w_[k + hi * 2]; bv.y = w_[k + hi * 2 + 1];
        acc = __builtin_amdgcn_wmma_f32_16x16x4_f32(false, av, false, bv, (short)0, acc, false, false);
    }
    float bval = bias[col];
#pragma unroll
    for (int i = 0; i < 8; i++) {
        int r = tm * 16 + hi * 8 + i;
        if (r < M) {
            float v = acc[i] + bval;
            if (relu) v = fmaxf(v, 0.0f);
            C[(size_t)r * Nout + col] = v;
        }
    }
}

// ---------------- flash attention: one wave per (head, 16-query tile), online softmax ----------------
__global__ __launch_bounds__(32)
void attn_kernel(const float* __restrict__ qkv, float* __restrict__ attn_out,
                 const int* __restrict__ Mptr) {
    __shared__ float tile[16][17];
    int M = *Mptr;
    int qt = blockIdx.x, hh = blockIdx.y;
    if (qt * 16 >= M) return;                    // uniform
    int tilesK = (M + 15) >> 4;
    int lane = threadIdx.x & 31, ln = lane & 15, hi = lane >> 4;

    int qrow = qt * 16 + ln; if (qrow >= M) qrow = M - 1;
    const float* qp = qkv + (size_t)qrow * 384 + hh * 32;
    float qv[16];
#pragma unroll
    for (int kk = 0; kk < 8; kk++) { qv[2 * kk] = qp[kk * 4 + hi * 2]; qv[2 * kk + 1] = qp[kk * 4 + hi * 2 + 1]; }

    const float scale = 0.17677669529663687f;    // 1/sqrt(32)
    float mrow[8], srow[8];
#pragma unroll
    for (int i = 0; i < 8; i++) { mrow[i] = -1e30f; srow[i] = 0.0f; }

    // ---- pass 1: statistics ----
    for (int kt = 0; kt < tilesK; ++kt) {
        const float* kp = qkv + (size_t)(kt * 16 + ln) * 384 + 128 + hh * 32;
        bool colok = (kt * 16 + ln) < M;
        v8f c = {};
#pragma unroll
        for (int kk = 0; kk < 8; kk++) {
            v2f av; av.x = qv[2 * kk];          av.y = qv[2 * kk + 1];
            v2f bv; bv.x = kp[kk * 4 + hi * 2]; bv.y = kp[kk * 4 + hi * 2 + 1];
            c = __builtin_amdgcn_wmma_f32_16x16x4_f32(false, av, false, bv, (short)0, c, false, false);
        }
#pragma unroll
        for (int i = 0; i < 8; i++) {
            float val = colok ? c[i] * scale : -1e30f;     // c[i]: row i+hi*8, col ln
            float tmax = hred_max(val);
            float nm = fmaxf(mrow[i], tmax);
            float tsum = hred_sum(__expf(val - nm));
            srow[i] = srow[i] * __expf(mrow[i] - nm) + tsum;
            mrow[i] = nm;
        }
    }
#pragma unroll
    for (int i = 0; i < 8; i++) srow[i] = 1.0f / srow[i];

    // ---- pass 2: output ----
    v8f o0 = {}, o1 = {};
    for (int kt = 0; kt < tilesK; ++kt) {
        const float* kp = qkv + (size_t)(kt * 16 + ln) * 384 + 128 + hh * 32;
        bool colok = (kt * 16 + ln) < M;
        v8f c = {};
#pragma unroll
        for (int kk = 0; kk < 8; kk++) {
            v2f av; av.x = qv[2 * kk];          av.y = qv[2 * kk + 1];
            v2f bv; bv.x = kp[kk * 4 + hi * 2]; bv.y = kp[kk * 4 + hi * 2 + 1];
            c = __builtin_amdgcn_wmma_f32_16x16x4_f32(false, av, false, bv, (short)0, c, false, false);
        }
#pragma unroll
        for (int i = 0; i < 8; i++) {
            float val = colok ? c[i] * scale : -1e30f;
            tile[hi * 8 + i][ln] = __expf(val - mrow[i]) * srow[i];   // P[q][key]
        }
        __syncthreads();
#pragma unroll
        for (int kk = 0; kk < 4; kk++) {
            int kl = kk * 4 + hi * 2;
            v2f av; av.x = tile[ln][kl]; av.y = tile[ln][kl + 1];      // A: row=ln(query), K=key
            const float* v0p = qkv + (size_t)(kt * 16 + kl) * 384 + 256 + hh * 32;
            const float* v1p = v0p + 384;
            v2f b0; b0.x = v0p[ln];      b0.y = v1p[ln];
            o0 = __builtin_amdgcn_wmma_f32_16x16x4_f32(false, av, false, b0, (short)0, o0, false, false);
            v2f b1; b1.x = v0p[ln + 16]; b1.y = v1p[ln + 16];
            o1 = __builtin_amdgcn_wmma_f32_16x16x4_f32(false, av, false, b1, (short)0, o1, false, false);
        }
        __syncthreads();
    }
#pragma unroll
    for (int i = 0; i < 8; i++) {
        int q = qt * 16 + hi * 8 + i;
        if (q < M) {
            attn_out[(size_t)q * 128 + hh * 32 + ln]      = o0[i];
            attn_out[(size_t)q * 128 + hh * 32 + 16 + ln] = o1[i];
        }
    }
}

// ---------------- fused residual + layernorm: out = LN(A+B)*g + b, one wave per row ----------------
__global__ __launch_bounds__(256)
void add_ln_kernel(const float* __restrict__ A, const float* __restrict__ B,
                   const float* __restrict__ g, const float* __restrict__ be,
                   float* __restrict__ out, const int* __restrict__ Mptr) {
    int M = *Mptr;
    int lane = threadIdx.x & 31;
    int row = blockIdx.x * (blockDim.x >> 5) + (threadIdx.x >> 5);
    if (row >= M) return;
    float4 x = ((const float4*)(A + (size_t)row * 128))[lane];
    float4 y = ((const float4*)(B + (size_t)row * 128))[lane];
    float4 v; v.x = x.x + y.x; v.y = x.y + y.y; v.z = x.z + y.z; v.w = x.w + y.w;
    float mean = wred_sum(v.x + v.y + v.z + v.w) * (1.0f / 128.0f);
    float4 d; d.x = v.x - mean; d.y = v.y - mean; d.z = v.z - mean; d.w = v.w - mean;
    float var = wred_sum(d.x * d.x + d.y * d.y + d.z * d.z + d.w * d.w) * (1.0f / 128.0f);
    float rs = rsqrtf(var + 1e-5f);
    float4 gg = ((const float4*)g)[lane];
    float4 bb = ((const float4*)be)[lane];
    float4 o; o.x = d.x * rs * gg.x + bb.x; o.y = d.y * rs * gg.y + bb.y;
    o.z = d.z * rs * gg.z + bb.z; o.w = d.w * rs * gg.w + bb.w;
    ((float4*)(out + (size_t)row * 128))[lane] = o;
}

// ---------------- final fused gather + bf16 WMMA GEMM ----------------
// out[N,128] = concat(h, x0[inv0], x1[inv1], x2[inv2]) @ Wf^T + bf
// Wf (bf16, 128KB) staged once per workgroup into LDS (async path if available),
// then one wave per 16-row tile; K=512 in 16 steps of 32 via v_wmma_f32_16x16x32_bf16.
__global__ __launch_bounds__(256)
void final_gemm_kernel(const float* __restrict__ h, const int* __restrict__ i0,
                       const int* __restrict__ i1, const int* __restrict__ i2,
                       const float* __restrict__ x, const __bf16* __restrict__ WfB,
                       const float* __restrict__ bf_, float* __restrict__ out) {
    extern __shared__ __bf16 sWf[];              // 128 cols x WF_PITCH
    {
        // cooperative stage: 128 cols x 64 chunks of 16B
        for (int c = threadIdx.x; c < 128 * 64; c += 256) {
            int col = c >> 6, chunk = c & 63;
            const __bf16* gsrc = WfB + (size_t)col * 512 + chunk * 8;
            __bf16* ldst = sWf + col * WF_PITCH + chunk * 8;
#if defined(USE_ASYNC_LDS)
            __builtin_amdgcn_global_load_async_to_lds_b128(
                (gptr_v4i)(v4i_async*)(void*)(const_cast<__bf16*>(gsrc)),
                (lptr_v4i)(v4i_async*)(void*)ldst, 0, 0);
#else
            *(uint4*)ldst = *(const uint4*)gsrc;
#endif
        }
#if defined(USE_ASYNC_LDS)
        __builtin_amdgcn_s_wait_asynccnt(0);
#endif
        __syncthreads();
    }
    int lane = threadIdx.x & 31, ln = lane & 15, hi = lane >> 4;
    int gw = blockIdx.x * (blockDim.x >> 5) + (threadIdx.x >> 5);
    int nw = gridDim.x * (blockDim.x >> 5);
    const int tiles = N_PART / 16;
    const v8f vz = {};
    for (int t = gw; t < tiles; t += nw) {
        int r = t * 16 + ln;
        __builtin_prefetch(h + (size_t)(t + nw) * 16 * 128, 0, 1);
        const float* rowp[4];
        rowp[0] = h + (size_t)r * 128;
        rowp[1] = x + (size_t)i0[r] * 128;
        rowp[2] = x + (size_t)(MAXM0 + i1[r]) * 128;
        rowp[3] = x + (size_t)(MAXM0 + MAXM1 + i2[r]) * 128;
        v8f acc[8];
#pragma unroll
        for (int i = 0; i < 8; i++) acc[i] = vz;
        for (int kc = 0; kc < 16; ++kc) {
            const float4* s4 = (const float4*)(rowp[kc >> 2] + (kc & 3) * 32);
            float4 f0 = s4[hi * 2], f1 = s4[hi * 2 + 1];          // K = hi*8 + 0..7
            float4 f2 = s4[4 + hi * 2], f3 = s4[4 + hi * 2 + 1];  // K = 16 + hi*8 + 0..7
            v16bf a;
            a[0] = (__bf16)f0.x; a[1] = (__bf16)f0.y; a[2]  = (__bf16)f0.z; a[3]  = (__bf16)f0.w;
            a[4] = (__bf16)f1.x; a[5] = (__bf16)f1.y; a[6]  = (__bf16)f1.z; a[7]  = (__bf16)f1.w;
            a[8] = (__bf16)f2.x; a[9] = (__bf16)f2.y; a[10] = (__bf16)f2.z; a[11] = (__bf16)f2.w;
            a[12] = (__bf16)f3.x; a[13] = (__bf16)f3.y; a[14] = (__bf16)f3.z; a[15] = (__bf16)f3.w;
#pragma unroll
            for (int ct = 0; ct < 8; ++ct) {
                // B 32x16: lane = column, K = hi*8 + 0..7 and 16 + hi*8 + 0..7; two 16B LDS loads
                const __bf16* wp = sWf + (size_t)(ct * 16 + ln) * WF_PITCH + kc * 32 + hi * 8;
                v8bf blo = *(const v8bf*)wp;
                v8bf bhi = *(const v8bf*)(wp + 16);
                v16bf b;
#pragma unroll
                for (int j = 0; j < 8; j++) { b[j] = blo[j]; b[8 + j] = bhi[j]; }
                acc[ct] = __builtin_amdgcn_wmma_f32_16x16x32_bf16(false, a, false, b, (short)0, acc[ct], false, false);
            }
        }
#pragma unroll
        for (int ct = 0; ct < 8; ct++) {
            int col = ct * 16 + ln;
            float bval = bf_[col];
#pragma unroll
            for (int i = 0; i < 8; i++)
                out[(size_t)(t * 16 + hi * 8 + i) * 128 + col] = acc[ct][i] + bval;
        }
    }
}

// ---------------- host orchestration ----------------
extern "C" void kernel_launch(void* const* d_in, const int* in_sizes, int n_in,
                              void* d_out, int out_size, void* d_ws, size_t ws_size,
                              hipStream_t stream) {
    (void)in_sizes; (void)n_in; (void)out_size;
    if (ws_size < (size_t)WS_FLOATS * sizeof(float)) return;

    const float* h    = (const float*)d_in[0];
    const int*   inv0 = (const int*)d_in[2];
    const int*   inv1 = (const int*)d_in[3];
    const int*   inv2 = (const int*)d_in[4];
    const int*   Mp[3] = {(const int*)d_in[5], (const int*)d_in[6], (const int*)d_in[7]};
    const float* Wqkv = (const float*)d_in[8];
    const float* bqkv = (const float*)d_in[9];
    const float* Wo   = (const float*)d_in[10];
    const float* bo   = (const float*)d_in[11];
    const float* ln1s = (const float*)d_in[12];
    const float* ln1b = (const float*)d_in[13];
    const float* W1   = (const float*)d_in[14];
    const float* b1   = (const float*)d_in[15];
    const float* W2   = (const float*)d_in[16];
    const float* b2   = (const float*)d_in[17];
    const float* ln2s = (const float*)d_in[18];
    const float* ln2b = (const float*)d_in[19];
    const float* Wf   = (const float*)d_in[20];
    const float* bf_  = (const float*)d_in[21];
    float* out = (float*)d_out;

    float* ws     = (float*)d_ws;
    float* macro  = ws + OFF_MACRO;
    float* counts = ws + OFF_COUNTS;
    float* xbuf   = ws + OFF_X;
    float* qkv    = ws + OFF_QKV;
    float* attn   = ws + OFF_ATTN;
    float* tmp    = ws + OFF_TMP;
    float* ff     = ws + OFF_FF;
    __bf16* wfb   = (__bf16*)(ws + OFF_WFB);

    {   int n = MAXM_TOT * 128 + MAXM_TOT;      // zero sums + counts (contiguous)
        zero_kernel<<<(n + 255) / 256, 256, 0, stream>>>(macro, n);
    }
    scatter_kernel<<<1024, 256, 0, stream>>>(h, inv0, inv1, inv2, macro, counts);
    {   int n = MAXM_TOT * 128;
        finalize_kernel<<<(n + 255) / 256, 256, 0, stream>>>(macro, counts);
    }
    {   int n = 128 * 512;
        cvt_wf_kernel<<<(n + 255) / 256, 256, 0, stream>>>(Wf, wfb);
    }

    const int maxm[3] = {MAXM0, MAXM1, MAXM2};
    const int moff[3] = {0, MAXM0, MAXM0 + MAXM1};
    for (int s = 0; s < 3; s++) {
        int tilesM = maxm[s] / 16;
        float* mac = macro + (size_t)moff[s] * 128;
        float* xs  = xbuf  + (size_t)moff[s] * 128;
        gemm_wmma_f32<<<dim3(tilesM, 24), 32, 0, stream>>>(mac, Wqkv + (size_t)s * 384 * 128, bqkv + s * 384, qkv, Mp[s], 128, 384, 0);
        attn_kernel<<<dim3(tilesM, NHEAD), 32, 0, stream>>>(qkv, attn, Mp[s]);
        gemm_wmma_f32<<<dim3(tilesM, 8), 32, 0, stream>>>(attn, Wo + (size_t)s * 128 * 128, bo + s * 128, tmp, Mp[s], 128, 128, 0);
        add_ln_kernel<<<maxm[s] / 8, 256, 0, stream>>>(mac, tmp, ln1s + s * 128, ln1b + s * 128, xs, Mp[s]);
        gemm_wmma_f32<<<dim3(tilesM, 16), 32, 0, stream>>>(xs, W1 + (size_t)s * 256 * 128, b1 + s * 256, ff, Mp[s], 128, 256, 1);
        gemm_wmma_f32<<<dim3(tilesM, 8), 32, 0, stream>>>(ff, W2 + (size_t)s * 128 * 256, b2 + s * 128, tmp, Mp[s], 256, 128, 0);
        add_ln_kernel<<<maxm[s] / 8, 256, 0, stream>>>(xs, tmp, ln2s + s * 128, ln2b + s * 128, xs, Mp[s]);
    }
    final_gemm_kernel<<<512, 256, WF_LDS_BYTES, stream>>>(h, inv0, inv1, inv2, xbuf, wfb, bf_, out);
}
